// SelfAttention_58703613002114
// MI455X (gfx1250) — compile-verified, matrix operand order checked by reading, not run on hardware
//
#include <hip/hip_runtime.h>
#include <stdint.h>

// ---------------------------------------------------------------------------
// Problem constants (fixed by the reference)
// ---------------------------------------------------------------------------
constexpr int BATCH = 2;
constexpr int SEQ   = 2048;
constexpr int DIMSZ = 1024;
constexpr int NH    = 16;
constexpr int HD    = 64;
constexpr int MROWS = BATCH * SEQ;   // 4096 token rows

// ---------------------------------------------------------------------------
// WMMA plumbing (gfx1250 / CDNA5, wave32)
// ---------------------------------------------------------------------------
typedef __attribute__((ext_vector_type(16))) __bf16 v16bf;
typedef __attribute__((ext_vector_type(8)))  float  v8f;
typedef __attribute__((ext_vector_type(4)))  int    v4i;

#define DEV static __device__ __forceinline__
#define GAS __attribute__((address_space(1)))
#define LAS __attribute__((address_space(3)))

#if __has_builtin(__builtin_amdgcn_global_load_async_to_lds_b128)
#define HAVE_ASYNC_LDS 1
#else
#define HAVE_ASYNC_LDS 0
#endif

DEV __bf16 f2bf(float f) {            // RNE float -> bf16
  union { float f; uint32_t u; } a; a.f = f;
  uint32_t r = a.u + 0x7fffu + ((a.u >> 16) & 1u);
  uint16_t h = (uint16_t)(r >> 16);
  __bf16 o; __builtin_memcpy(&o, &h, 2); return o;
}

union Frag {                          // 16 bf16 per lane = 8 VGPRs
  v16bf v;
  uint4 q[2];
};

DEV v8f wmma_bf16(const Frag& a, const Frag& b, v8f c) {
  // (neg_a, A, neg_b, B, c_mod, C, reuse_a, reuse_b)
  return __builtin_amdgcn_wmma_f32_16x16x32_bf16(false, a.v, false, b.v,
                                                 (short)0, c, false, false);
}

// A/B 16-bit operand layout (ISA 7.12.2): lane half lh=lane>>4 selects the
// K sub-ranges; element i -> K = (i>>3)*16 + lh*8 + (i&7).  So a fragment is
// two 16-byte loads at K-offsets {lh*8, 16+lh*8} from a K-contiguous row.
DEV void load_frag(Frag& f, const __bf16* rowK, int lh) {
  f.q[0] = *(const uint4*)(rowK + lh * 8);
  f.q[1] = *(const uint4*)(rowK + 16 + lh * 8);
}

#if HAVE_ASYNC_LDS
// Per-lane 16-byte global -> LDS copy, tracked with ASYNCcnt.
DEV void async_copy_b128(const void* gsrc, void* ldst) {
  __builtin_amdgcn_global_load_async_to_lds_b128(
      (GAS v4i*)gsrc, (LAS v4i*)ldst, 0, 0);
}
#endif

// ---------------------------------------------------------------------------
// Elementwise conversion / transpose kernels
// ---------------------------------------------------------------------------
__global__ __launch_bounds__(256) void cvt_f32_bf16(
    const float* __restrict__ in, __bf16* __restrict__ out, int n) {
  int i = blockIdx.x * 256 + threadIdx.x;
  if (i < n) out[i] = f2bf(in[i]);
}

// in: rows_in x cols_in (f32 row-major)  ->  out: cols_in x rows_in (bf16)
__global__ __launch_bounds__(256) void transpose_cvt(
    const float* __restrict__ in, __bf16* __restrict__ out,
    int rows_in, int cols_in) {
  int i = blockIdx.x * 256 + threadIdx.x;
  if (i >= rows_in * cols_in) return;
  int n = i / rows_in;          // out row
  int k = i - n * rows_in;      // out col
  out[(size_t)n * rows_in + k] = f2bf(in[(size_t)k * cols_in + n]);
}

// ---------------------------------------------------------------------------
// Tiled "TN" GEMM:  C[M,N] (f32, +opt bias) = A[M,K](bf16) x Bt[N,K](bf16)^T
// Block: 64x128, BK=32, 8 waves -> each wave a 32x32 tile (2x2 WMMAs).
// Operand staging uses GLOBAL_LOAD_ASYNC_TO_LDS_B128 (ASYNCcnt) when the
// toolchain exposes it; otherwise classic global->VGPR->LDS.
// ---------------------------------------------------------------------------
template <bool HAS_BIAS>
__global__ __launch_bounds__(256) void gemm_tn_bf16(
    const __bf16* __restrict__ A, const __bf16* __restrict__ Bt,
    const float* __restrict__ bias, float* __restrict__ C,
    int M, int N, int K) {
  constexpr int BM = 64, BNT = 128, BK = 32, LDS_STRIDE = 40; // 80B rows
  __shared__ __attribute__((aligned(16))) __bf16 sA[BM  * LDS_STRIDE];
  __shared__ __attribute__((aligned(16))) __bf16 sB[BNT * LDS_STRIDE];

  const int t = threadIdx.x;
  const int lane = t & 31, wave = t >> 5;
  const int lane15 = lane & 15, lh = lane >> 4;
  const int wr = wave >> 2, wc = wave & 3;          // 2 x 4 wave grid
  const int m0 = blockIdx.y * BM, n0 = blockIdx.x * BNT;

  v8f acc[2][2] = {};

  const int arow = t >> 2, aseg = t & 3;            // A: 64 rows x 4 uint4

  for (int k0 = 0; k0 < K; k0 += BK) {
#if HAVE_ASYNC_LDS
    __syncthreads();  // protect previous iteration's LDS reads
    async_copy_b128(A + (size_t)(m0 + arow) * K + k0 + aseg * 8,
                    sA + arow * LDS_STRIDE + aseg * 8);
#pragma unroll
    for (int it = 0; it < 2; ++it) {
      int idx = t + it * 256;
      int brow = idx >> 2, bseg = idx & 3;
      async_copy_b128(Bt + (size_t)(n0 + brow) * K + k0 + bseg * 8,
                      sB + brow * LDS_STRIDE + bseg * 8);
    }
    asm volatile("s_wait_asynccnt 0" ::: "memory");
    __syncthreads();
#else
    uint4 aR = *(const uint4*)(A + (size_t)(m0 + arow) * K + k0 + aseg * 8);
    uint4 bR[2];
#pragma unroll
    for (int it = 0; it < 2; ++it) {
      int idx = t + it * 256;
      int brow = idx >> 2, bseg = idx & 3;
      bR[it] = *(const uint4*)(Bt + (size_t)(n0 + brow) * K + k0 + bseg * 8);
    }
    __syncthreads();
    *(uint4*)(sA + arow * LDS_STRIDE + aseg * 8) = aR;
#pragma unroll
    for (int it = 0; it < 2; ++it) {
      int idx = t + it * 256;
      int brow = idx >> 2, bseg = idx & 3;
      *(uint4*)(sB + brow * LDS_STRIDE + bseg * 8) = bR[it];
    }
    __syncthreads();
#endif

    Frag af[2], bfr[2];
#pragma unroll
    for (int i = 0; i < 2; ++i)
      load_frag(af[i], sA + (wr * 32 + i * 16 + lane15) * LDS_STRIDE, lh);
#pragma unroll
    for (int j = 0; j < 2; ++j)
      load_frag(bfr[j], sB + (wc * 32 + j * 16 + lane15) * LDS_STRIDE, lh);

#pragma unroll
    for (int i = 0; i < 2; ++i)
#pragma unroll
      for (int j = 0; j < 2; ++j)
        acc[i][j] = wmma_bf16(af[i], bfr[j], acc[i][j]);

#if !HAVE_ASYNC_LDS
    __syncthreads();  // protect this iteration's reads before next overwrite
#endif
  }

  // D layout: VGPR r -> row 8*lh + r; col = lane15
#pragma unroll
  for (int i = 0; i < 2; ++i) {
#pragma unroll
    for (int j = 0; j < 2; ++j) {
      int col = n0 + wc * 32 + j * 16 + lane15;
      float bv = HAS_BIAS ? bias[col] : 0.f;
#pragma unroll
      for (int r = 0; r < 8; ++r) {
        int row = m0 + wr * 32 + i * 16 + 8 * lh + r;
        C[(size_t)row * N + col] = acc[i][j][r] + bv;
      }
    }
  }
}

// ---------------------------------------------------------------------------
// RoPE + head repack.  qkv: (B*N, 3*DIM) f32.
//   qh,kh: (B*H, N, 64) bf16 (roped);  vt: (B*H, 64, N) bf16 (transposed).
// One thread per (b,n,h,pair j in [0,32)).
// ---------------------------------------------------------------------------
__global__ __launch_bounds__(256) void rope_pack(
    const float* __restrict__ qkv, const float* __restrict__ pos,
    __bf16* __restrict__ qh, __bf16* __restrict__ kh,
    __bf16* __restrict__ vt) {
  int idx = blockIdx.x * 256 + threadIdx.x;
  int j = idx & 31;
  int h = (idx >> 5) & (NH - 1);
  int n = (idx >> 9) & (SEQ - 1);
  int b = idx >> 20;
  if (b >= BATCH) return;
  int bh = b * NH + h;

  int p = j >> 4, fi = j & 15;                       // axis / freq index
  float freq  = __expf((float)fi * -0.5756462732f);  // 10000^(-fi/16)
  float theta = pos[((size_t)b * SEQ + n) * 2 + p] * freq;
  float sn, cs;
  __sincosf(theta, &sn, &cs);

  const float* row = qkv + ((size_t)b * SEQ + n) * (3 * DIMSZ);
  {
    float x1 = row[h * HD + j], x2 = row[h * HD + 32 + j];
    __bf16* q = qh + ((size_t)bh * SEQ + n) * HD;
    q[j]      = f2bf(x1 * cs - x2 * sn);
    q[32 + j] = f2bf(x1 * sn + x2 * cs);
  }
  {
    float x1 = row[DIMSZ + h * HD + j], x2 = row[DIMSZ + h * HD + 32 + j];
    __bf16* k = kh + ((size_t)bh * SEQ + n) * HD;
    k[j]      = f2bf(x1 * cs - x2 * sn);
    k[32 + j] = f2bf(x1 * sn + x2 * cs);
  }
  {
    float v1 = row[2 * DIMSZ + h * HD + j];
    float v2 = row[2 * DIMSZ + h * HD + 32 + j];
    vt[((size_t)bh * HD + j)      * SEQ + n] = f2bf(v1);
    vt[((size_t)bh * HD + 32 + j) * SEQ + n] = f2bf(v2);
  }
}

// ---------------------------------------------------------------------------
// Flash attention.  Grid: (B*H, SEQ/128); 8 waves/block, one wave = 16 queries
// over all 2048 keys.  Key step = 64 so the cross-lane softmax reductions
// (8 rows x 4 xor-steps x {max,sum}) are amortized over 16 WMMAs per round.
// attnb: (B, N, H*64) bf16.
// ---------------------------------------------------------------------------
__global__ __launch_bounds__(256) void flash_attn(
    const __bf16* __restrict__ qh, const __bf16* __restrict__ kh,
    const __bf16* __restrict__ vt, __bf16* __restrict__ attnb) {
  __shared__ __attribute__((aligned(16))) __bf16 sP[8 * 16 * 64]; // per-wave P
  const int t = threadIdx.x, lane = t & 31, wave = t >> 5;
  const int lane15 = lane & 15, lh = lane >> 4;
  const int bh = blockIdx.x;
  const int b = bh >> 4, h = bh & (NH - 1);
  const int q0 = blockIdx.y * 128 + wave * 16;

  // Q fragments (contraction d split 0..31 / 32..63), kept resident
  Frag qf[2];
  {
    const __bf16* qp = qh + ((size_t)bh * SEQ + q0 + lane15) * HD;
    load_frag(qf[0], qp, lh);
    load_frag(qf[1], qp + 32, lh);
  }

  v8f o[4] = {};
  float mrun[8], lrun[8];
#pragma unroll
  for (int r = 0; r < 8; ++r) { mrun[r] = -3.0e38f; lrun[r] = 0.f; }

  __bf16* myP = sP + wave * (16 * 64);

  for (int k0 = 0; k0 < SEQ; k0 += 64) {
    // ---- S = (Q K^T) * scale for 4 x 16-key sub-tiles ----
    v8f s[4];
#pragma unroll
    for (int kt = 0; kt < 4; ++kt) {
      Frag kf0, kf1;
      const __bf16* kp = kh + ((size_t)bh * SEQ + k0 + kt * 16 + lane15) * HD;
      load_frag(kf0, kp, lh);
      load_frag(kf1, kp + 32, lh);
      v8f z = {};
      z = wmma_bf16(qf[0], kf0, z);
      z = wmma_bf16(qf[1], kf1, z);
      s[kt] = z * 0.125f;                       // 1/sqrt(64)
    }

    // prefetch next round's K/V tiles toward cache (global_prefetch_b8)
    if (k0 + 64 < SEQ) {
      __builtin_prefetch(kh + ((size_t)bh * SEQ + k0 + 64 + lane) * HD, 0, 0);
      __builtin_prefetch(
          vt + ((size_t)bh * HD + (lane & 15) * 4) * SEQ + k0 + 64, 0, 0);
    }

    // ---- online softmax stats: row = query 8*lh + r, reduce 16 lanes ----
    float mnew[8], alpha[8];
#pragma unroll
    for (int r = 0; r < 8; ++r) {
      float tm = fmaxf(fmaxf(s[0][r], s[1][r]), fmaxf(s[2][r], s[3][r]));
#pragma unroll
      for (int off = 1; off < 16; off <<= 1)
        tm = fmaxf(tm, __shfl_xor(tm, off, 32));
      mnew[r]  = fmaxf(mrun[r], tm);
      alpha[r] = __expf(mrun[r] - mnew[r]);
      mrun[r]  = mnew[r];
      lrun[r] *= alpha[r];
    }
#pragma unroll
    for (int tt = 0; tt < 4; ++tt)
#pragma unroll
      for (int r = 0; r < 8; ++r) o[tt][r] *= alpha[r];

#pragma unroll
    for (int r = 0; r < 8; ++r) {
      float p0 = __expf(s[0][r] - mnew[r]);
      float p1 = __expf(s[1][r] - mnew[r]);
      float p2 = __expf(s[2][r] - mnew[r]);
      float p3 = __expf(s[3][r] - mnew[r]);
      s[0][r] = p0; s[1][r] = p1; s[2][r] = p2; s[3][r] = p3;
      float rs = (p0 + p1) + (p2 + p3);
#pragma unroll
      for (int off = 1; off < 16; off <<= 1)
        rs += __shfl_xor(rs, off, 32);
      lrun[r] += rs;
    }

    // ---- stage P (16 queries x 64 keys) to LDS, re-read in A layout ----
#pragma unroll
    for (int kt = 0; kt < 4; ++kt)
#pragma unroll
      for (int r = 0; r < 8; ++r)
        myP[(8 * lh + r) * 64 + kt * 16 + lane15] = f2bf(s[kt][r]);
    asm volatile("s_wait_dscnt 0" ::: "memory");  // same-wave DS ordering

    Frag pf[2];
    load_frag(pf[0], myP + lane15 * 64, lh);        // keys k0 .. k0+31
    load_frag(pf[1], myP + lane15 * 64 + 32, lh);   // keys k0+32 .. k0+63

    // ---- O += P @ V over the 4 head-dim tiles ----
#pragma unroll
    for (int tt = 0; tt < 4; ++tt) {
      Frag vf0, vf1;
      const __bf16* vp =
          vt + ((size_t)bh * HD + tt * 16 + lane15) * SEQ + k0;
      load_frag(vf0, vp, lh);
      load_frag(vf1, vp + 32, lh);
      o[tt] = wmma_bf16(pf[0], vf0, o[tt]);
      o[tt] = wmma_bf16(pf[1], vf1, o[tt]);
    }
    asm volatile("" ::: "memory");
  }

  // ---- normalize and write (B, N, H*64) bf16 ----
#pragma unroll
  for (int r = 0; r < 8; ++r) {
    float inv = (lrun[r] > 0.f) ? 1.f / lrun[r] : 0.f;
    int row = q0 + 8 * lh + r;
    __bf16* op = attnb + ((size_t)b * SEQ + row) * DIMSZ + h * HD;
#pragma unroll
    for (int tt = 0; tt < 4; ++tt)
      op[tt * 16 + lane15] = f2bf(o[tt][r] * inv);
  }
}

// ---------------------------------------------------------------------------
// Host-side orchestration
// ---------------------------------------------------------------------------
extern "C" void kernel_launch(void* const* d_in, const int* in_sizes, int n_in,
                              void* d_out, int out_size, void* d_ws,
                              size_t ws_size, hipStream_t stream) {
  (void)in_sizes; (void)n_in; (void)out_size; (void)ws_size;
  const float* x     = (const float*)d_in[0];
  const float* pos   = (const float*)d_in[1];
  const float* w_qkv = (const float*)d_in[2];
  const float* w_out = (const float*)d_in[3];
  const float* b_out = (const float*)d_in[4];
  float* out = (float*)d_out;

  char* ws = (char*)d_ws;
  size_t off = 0;
  auto take = [&](size_t bytes) -> char* {
    char* p = ws + off;
    off += (bytes + 255) & ~(size_t)255;
    return p;
  };

  __bf16* xb    = (__bf16*)take((size_t)MROWS * DIMSZ * 2);          //  8 MB
  __bf16* wqkvT = (__bf16*)take((size_t)3 * DIMSZ * DIMSZ * 2);      //  6 MB
  __bf16* woutT = (__bf16*)take((size_t)DIMSZ * DIMSZ * 2);          //  2 MB
  float*  qkv   = (float*) take((size_t)MROWS * 3 * DIMSZ * 4);      // 48 MB
  __bf16* qhp   = (__bf16*)take((size_t)BATCH * NH * SEQ * HD * 2);  //  8 MB
  __bf16* khp   = (__bf16*)take((size_t)BATCH * NH * SEQ * HD * 2);  //  8 MB
  __bf16* vtp   = (__bf16*)take((size_t)BATCH * NH * SEQ * HD * 2);  //  8 MB
  __bf16* attnb = (__bf16*)take((size_t)MROWS * DIMSZ * 2);          //  8 MB

  // 1) precision staging (bf16; weights transposed for TN-GEMM K-contiguity)
  cvt_f32_bf16<<<(MROWS * DIMSZ + 255) / 256, 256, 0, stream>>>(
      x, xb, MROWS * DIMSZ);
  transpose_cvt<<<(DIMSZ * 3 * DIMSZ + 255) / 256, 256, 0, stream>>>(
      w_qkv, wqkvT, DIMSZ, 3 * DIMSZ);
  transpose_cvt<<<(DIMSZ * DIMSZ + 255) / 256, 256, 0, stream>>>(
      w_out, woutT, DIMSZ, DIMSZ);

  // 2) qkv = x @ w_qkv          (4096 x 3072, K=1024)
  gemm_tn_bf16<false><<<dim3(3 * DIMSZ / 128, MROWS / 64), 256, 0, stream>>>(
      xb, wqkvT, nullptr, qkv, MROWS, 3 * DIMSZ, DIMSZ);

  // 3) axial RoPE + head repack (V transposed for PV WMMA B-operand)
  rope_pack<<<(BATCH * SEQ * NH * 32) / 256, 256, 0, stream>>>(
      qkv, pos, qhp, khp, vtp);

  // 4) flash attention
  flash_attn<<<dim3(BATCH * NH, SEQ / 128), 256, 0, stream>>>(
      qhp, khp, vtp, attnb);

  // 5) out = attn @ w_out + b   (4096 x 1024, K=1024)
  gemm_tn_bf16<true><<<dim3(DIMSZ / 128, MROWS / 64), 256, 0, stream>>>(
      attnb, woutT, b_out, out, MROWS, DIMSZ, DIMSZ);
}
